// DifferentiableRenderer_3204045603837
// MI455X (gfx1250) — matrix operand order repaired and checked by reading, not compiled, not run on hardware
//
#include <hip/hip_runtime.h>
#include <stdint.h>

typedef __attribute__((ext_vector_type(16))) _Float16 v16h;
typedef __attribute__((ext_vector_type(8)))  float    v8f;

#define N_GAUSS 4096
#define N_CHUNK (N_GAUSS / 32)
#define HW      128
#define NPIX    (HW * HW)
#define LOG2E   1.4426950408889634f

// ---------------------------------------------------------------------------
// Kernel 1: per (camera, gaussian) projection + exp-scale precompute.
// params[cam*N + n] = { proj_x, proj_y, -0.5*log2(e)/var, opacity }
// ---------------------------------------------------------------------------
__global__ void dr_preprocess(const float* __restrict__ positions,
                              const float* __restrict__ opacities,
                              const float* __restrict__ scales,
                              const float* __restrict__ qvec,
                              const float* __restrict__ tvec,
                              float4* __restrict__ params,
                              int nTotal, int N)
{
    int t = blockIdx.x * blockDim.x + threadIdx.x;
    if (t >= nTotal) return;
    int cam = t / N;
    int n   = t - cam * N;

    float qw = qvec[cam * 4 + 0];
    float qx = qvec[cam * 4 + 1];
    float qy = qvec[cam * 4 + 2];
    float qz = qvec[cam * 4 + 3];
    float inv = rsqrtf(qw * qw + qx * qx + qy * qy + qz * qz);
    qw *= inv; qx *= inv; qy *= inv; qz *= inv;

    float r00 = 1.0f - 2.0f * (qy * qy + qz * qz);
    float r01 = 2.0f * (qx * qy - qz * qw);
    float r02 = 2.0f * (qx * qz + qy * qw);
    float r10 = 2.0f * (qx * qy + qz * qw);
    float r11 = 1.0f - 2.0f * (qx * qx + qz * qz);
    float r12 = 2.0f * (qy * qz - qx * qw);
    float r20 = 2.0f * (qx * qz - qy * qw);
    float r21 = 2.0f * (qy * qz + qx * qw);
    float r22 = 1.0f - 2.0f * (qx * qx + qy * qy);

    float px = positions[n * 3 + 0];
    float py = positions[n * 3 + 1];
    float pz = positions[n * 3 + 2];
    float tx = tvec[cam * 3 + 0];
    float ty = tvec[cam * 3 + 1];
    float tz = tvec[cam * 3 + 2];

    float xc = r00 * px + r01 * py + r02 * pz + tx;
    float yc = r10 * px + r11 * py + r12 * pz + ty;
    float zc = r20 * px + r21 * py + r22 * pz + tz;

    float projx = xc / zc * 500.0f + 64.0f;   // FX, CX
    float projy = yc / zc * 500.0f + 64.0f;   // FY, CY

    float s   = scales[n];
    float var = s * s;
    float ce  = -0.5f * LOG2E / var;          // w = op * exp2(ce * dist2)

    params[t] = make_float4(projx, projy, ce, opacities[n]);
}

// ---------------------------------------------------------------------------
// Kernel 2: pre-pack augmented colors into WMMA A-fragment register layout.
// A is 16x32 f16 (M=channel row 0..2 = RGB, row 3 = ones, rows 4..15 = 0).
// 16-bit A layout: lane L holds row M=L%16; VGPR v holds packed K pair
//   k0 = 2v + (v>=4 ? 8 : 0) + (L>=16 ? 8 : 0).
// Output dword index: chunk*256 + lane*8 + v   (128 chunks -> 128 KB).
// One thread per packed dword.
// ---------------------------------------------------------------------------
__global__ void dr_pack_colors(const float* __restrict__ colors,
                               uint32_t* __restrict__ aFrag)
{
    int t = blockIdx.x * blockDim.x + threadIdx.x;   // 128*256 threads
    int chunk = t >> 8;
    int r     = t & 255;
    int lane  = r >> 3;
    int v     = r & 7;

    int hi = (lane >= 16) ? 1 : 0;
    int ch = lane & 15;
    int ak = 2 * v + (v >= 4 ? 8 : 0) + (hi ? 8 : 0);
    int n0 = chunk * 32 + ak;

    float f0, f1;
    if (ch < 3)      { f0 = colors[n0 * 3 + ch]; f1 = colors[(n0 + 1) * 3 + ch]; }
    else if (ch == 3){ f0 = 1.0f; f1 = 1.0f; }
    else             { f0 = 0.0f; f1 = 0.0f; }

    struct H2 { _Float16 lo, hi; } pk;
    pk.lo = (_Float16)f0;
    pk.hi = (_Float16)f1;
    aFrag[t] = __builtin_bit_cast(uint32_t, pk);
}

// ---------------------------------------------------------------------------
// Kernel 3: WMMA render.
//   D(16x16) = A(16x32 f16: aug-colors) x B(32x16 f16: weights, N=pixel)
//   accumulated in f32 over 128 chunks of 32 gaussians.
// 8 waves/block, 16 pixels/wave -> 128 pixels/block, 128 blocks/camera.
// ---------------------------------------------------------------------------
struct AFragRaw { uint4 q0, q1; };   // 32 bytes == v16h

__global__ __launch_bounds__(256)
void dr_render(const float4* __restrict__ params,
               const uint4*  __restrict__ aFrag4,
               float* __restrict__ out)
{
    __shared__ float4 sP[N_GAUSS];            // 64 KB: this camera's params

    int cam = blockIdx.x >> 7;                // 128 blocks per camera
    int blk = blockIdx.x & 127;

    const float4* camP = params + cam * N_GAUSS;
    for (int i = threadIdx.x; i < N_GAUSS; i += 256)
        sP[i] = camP[i];
    __syncthreads();

    int lane = threadIdx.x & 31;
    int wave = threadIdx.x >> 5;
    int hi   = (lane >= 16) ? 1 : 0;
    int col  = lane & 15;                     // B-matrix column (pixel)

    int p  = blk * 128 + wave * 16 + col;     // this lane's pixel
    float fx = (float)(p & (HW - 1));
    float fy = (float)(p >> 7);

    int aBase = lane * 2;                     // uint4 index within a chunk row

    v8f acc = {};

    for (int c = 0; c < N_CHUNK; ++c) {
        int g = c * 32;

        // ---- A fragment: precomputed packed colors, 2 coalesced b128 loads
        AFragRaw raw;
        raw.q0 = aFrag4[c * 64 + aBase];
        raw.q1 = aFrag4[c * 64 + aBase + 1];
        v16h aF = __builtin_bit_cast(v16h, raw);

        // ---- B fragment: weights, 16-bit B layout ----
        // lanes 0-15 hold K=0..7,16..23, lanes 16-31 hold K=8..15,24..31 (broadcast LDS)
        v16h bF;
#pragma unroll
        for (int v = 0; v < 8; ++v) {
            int bn0 = g + 2 * v + (v >= 4 ? 8 : 0) + (hi ? 8 : 0);
            float4 g0 = sP[bn0];
            float4 g1 = sP[bn0 + 1];

            float dx0 = fx - g0.x, dy0 = fy - g0.y;
            float d20 = dx0 * dx0 + dy0 * dy0;
            float w0  = g0.w * __builtin_amdgcn_exp2f(d20 * g0.z);

            float dx1 = fx - g1.x, dy1 = fy - g1.y;
            float d21 = dx1 * dx1 + dy1 * dy1;
            float w1  = g1.w * __builtin_amdgcn_exp2f(d21 * g1.z);

            bF[2 * v]     = (_Float16)w0;
            bF[2 * v + 1] = (_Float16)w1;
        }

        // D = A x B + C   (f32 accumulate)
        acc = __builtin_amdgcn_wmma_f32_16x16x32_f16(
                  false, aF, false, bF, (short)0, acc, false, false);
    }

    // D layout: VGPR r, lanes 0-15: M=r (channel), N=lane (pixel col).
    // Channels 0..2 = num RGB, channel 3 = den.
    if (lane < 16) {
        float den    = fmaxf(acc[3] + 1e-8f, 1e-8f);
        float invden = 1.0f / den;
        out[(cam * 3 + 0) * NPIX + p] = acc[0] * invden;
        out[(cam * 3 + 1) * NPIX + p] = acc[1] * invden;
        out[(cam * 3 + 2) * NPIX + p] = acc[2] * invden;
    }
}

// ---------------------------------------------------------------------------
extern "C" void kernel_launch(void* const* d_in, const int* in_sizes, int n_in,
                              void* d_out, int out_size, void* d_ws, size_t ws_size,
                              hipStream_t stream)
{
    const float* positions = (const float*)d_in[0];
    const float* colors    = (const float*)d_in[1];
    const float* opacities = (const float*)d_in[2];
    const float* scales    = (const float*)d_in[3];
    const float* qvec      = (const float*)d_in[4];
    const float* tvec      = (const float*)d_in[5];

    int N = in_sizes[2];          // 4096 gaussians
    int B = in_sizes[4] / 4;      // 2 cameras
    int nTotal = B * N;

    // Workspace layout:
    //   [0, 128 KB)    params: B*N float4
    //   [128, 256 KB)  aFrag:  N_CHUNK*32 lanes*8 dwords (packed f16 colors)
    float4*   params = (float4*)d_ws;
    uint32_t* aFrag  = (uint32_t*)((char*)d_ws + (size_t)B * N * sizeof(float4));

    dr_preprocess<<<(nTotal + 255) / 256, 256, 0, stream>>>(
        positions, opacities, scales, qvec, tvec, params, nTotal, N);

    dr_pack_colors<<<N_CHUNK, 256, 0, stream>>>(colors, aFrag);

    dr_render<<<B * 128, 256, 0, stream>>>(
        params, (const uint4*)aFrag, (float*)d_out);
}
